// KindredAttention_41532333753038
// MI455X (gfx1250) — matrix operand order repaired and checked by reading, not compiled, not run on hardware
//
#include <hip/hip_runtime.h>
#include <hip/hip_bf16.h>

// ---------- types ----------
typedef __attribute__((ext_vector_type(16))) __bf16 v16bf;
typedef __attribute__((ext_vector_type(8)))  float  v8f;
typedef __attribute__((ext_vector_type(4)))  unsigned int u32x4;
typedef __attribute__((ext_vector_type(8)))  unsigned int u32x8;

union Frag { uint4 u[2]; v16bf v; };   // 32 bytes: 16 bf16 per lane

#define S_LEN   2048
#define HEADS   16
#define HDIM    64
#define HID     1024
#define BATCH   4
#define MROWS   (BATCH * S_LEN)        // 8192

// ---------- helpers ----------
__device__ __forceinline__ unsigned short f2bf(float f) {
    unsigned int x = __float_as_uint(f);
    unsigned int r = (x + 0x7FFFu + ((x >> 16) & 1u)) >> 16;
    return (unsigned short)r;
}

__device__ __forceinline__ float rmax16(float v) {
    v = fmaxf(v, __shfl_xor(v, 1, 32));
    v = fmaxf(v, __shfl_xor(v, 2, 32));
    v = fmaxf(v, __shfl_xor(v, 4, 32));
    v = fmaxf(v, __shfl_xor(v, 8, 32));
    return v;
}
__device__ __forceinline__ float rsum16(float v) {
    v += __shfl_xor(v, 1, 32);
    v += __shfl_xor(v, 2, 32);
    v += __shfl_xor(v, 4, 32);
    v += __shfl_xor(v, 8, 32);
    return v;
}

__device__ __forceinline__ v8f wmma_bf16(v16bf a, v16bf b, v8f c) {
    return __builtin_amdgcn_wmma_f32_16x16x32_bf16(
        /*neg_a=*/false, a, /*neg_b=*/false, b,
        /*c_mod=*/(short)0, c, /*reuse_a=*/false, /*reuse_b=*/false);
}

// ---------- TDM: DMA a [rows x 32] bf16 panel (K-contiguous) into LDS ----------
// D# per CDNA5 ISA ch8: group0 = {count|flags, lds_addr, global_addr lo/hi|type},
// group1 = {mask|data_size, abar|dim0.lo, dim0.hi|dim1.lo, dim1.hi|tile0,
//           tile1|tile2, dim0_stride.lo, dim0_stride.hi|dim1_stride.lo, ...}
__device__ __forceinline__ void tdm_load_panel(const unsigned short* gsrc,
                                               unsigned lds_off_bytes,
                                               unsigned rows, unsigned K,
                                               unsigned tensor_rows) {
    unsigned long long ga = (unsigned long long)(uintptr_t)gsrc;
    u32x4 g0;
    g0[0] = 1u;                                    // count=1, user descriptor
    g0[1] = lds_off_bytes;                         // lds_addr (bytes)
    g0[2] = (unsigned)ga;                          // global_addr[31:0]
    g0[3] = ((unsigned)(ga >> 32) & 0x01FFFFFFu)   // global_addr[56:32]
            | (2u << 30);                          // type = 2 ("image")
    u32x8 g1;
    g1[0] = (1u << 16);                            // workgroup_mask=0, data_size=1 (2B)
    g1[1] = (K & 0xFFFFu) << 16;                   // tensor_dim0[15:0]
    g1[2] = ((K >> 16) & 0xFFFFu)                  // tensor_dim0[31:16]
            | ((tensor_rows & 0xFFFFu) << 16);     // tensor_dim1[15:0]
    g1[3] = ((tensor_rows >> 16) & 0xFFFFu)        // tensor_dim1[31:16]
            | (32u << 16);                         // tile_dim0 = 32 elems (64B)
    g1[4] = rows & 0xFFFFu;                        // tile_dim1 = rows, tile_dim2 = 0
    g1[5] = K;                                     // tensor_dim0_stride[31:0]
    g1[6] = 0u;                                    // stride hi / dim1_stride lo
    g1[7] = 0u;
    asm volatile("tensor_load_to_lds %0, %1" :: "s"(g0), "s"(g1) : "memory");
}

// ---------- fp32 -> bf16 conversion ----------
__global__ void cvt_bf16_kernel(const float* __restrict__ src,
                                unsigned short* __restrict__ dst, int n) {
    int idx = blockIdx.x * blockDim.x + threadIdx.x;
    int stride = gridDim.x * blockDim.x;
    for (int i = idx; i < n; i += stride) dst[i] = f2bf(src[i]);
}

// ---------- workgroup GEMM core: 4 waves, 64M x 256N block tile ----------
// Wave w computes the 64x64 sub-tile at n = n_block_base + w*64.
// W panel (256 rows x 32 K, bf16) is TDM-staged into LDS, double-buffered.
// A-frag (16x32 bf16): lane<16 row=lrow, K {0..7,16..23}; lane>=16 K {8..15,24..31}
// B-frag (32x16 bf16): lane<16 col=lrow K {0..7,16..23}; lane>=16 K {8..15,24..31}
__device__ __forceinline__ void gemm_block_tile(
    const unsigned short* __restrict__ A, const unsigned short* __restrict__ W,
    int K, int tensor_rows, int m_base, int n_block_base,
    int wid, int lid, unsigned short* ldsW /* 2 * 256*32 bf16 */,
    v8f acc[4][4]) {
    const int half = lid >> 4, lrow = lid & 15;
    const unsigned lds_base = (unsigned)(uintptr_t)ldsW;  // low 32 bits = LDS offset

    if (wid == 0)   // prologue: stage first K-slab into buffer 0
        tdm_load_panel(W + (size_t)n_block_base * K, lds_base, 256, K, tensor_rows);

    for (int kb = 0; kb < K; kb += 32) {
        const int buf = (kb >> 5) & 1;
        if (wid == 0) __builtin_amdgcn_s_wait_tensorcnt(0);  // current panel landed
        __syncthreads();                                     // visible to all waves
        if (wid == 0 && (kb + 32) < K)                       // overlap next DMA
            tdm_load_panel(W + (size_t)n_block_base * K + kb + 32,
                           lds_base + (unsigned)((buf ^ 1) * 16384), 256, K,
                           tensor_rows);

        Frag af[4], bf[4];
#pragma unroll
        for (int i = 0; i < 4; ++i) {
            const unsigned short* ab =
                A + (size_t)(m_base + 16 * i + lrow) * K + kb + half * 8;
            af[i].u[0] = *(const uint4*)(ab);
            af[i].u[1] = *(const uint4*)(ab + 16);
            __builtin_prefetch(ab + 32, 0, 3);   // global_prefetch_b8 next slab
        }
        const unsigned short* lw = ldsW + buf * 8192;
#pragma unroll
        for (int j = 0; j < 4; ++j) {
            const unsigned short* bb =
                lw + (wid * 64 + 16 * j + lrow) * 32 + half * 8;
            bf[j].u[0] = *(const uint4*)(bb);    // ds_load_b128
            bf[j].u[1] = *(const uint4*)(bb + 16);
        }
#pragma unroll
        for (int i = 0; i < 4; ++i)
#pragma unroll
            for (int j = 0; j < 4; ++j)
                acc[i][j] = wmma_bf16(af[i].v, bf[j].v, acc[i][j]);
    }
}

// ---------- QKV GEMM + fused RoPE epilogue ----------
// writes q,k as bf16 [B,H,S,D] (RoPE applied), v transposed as bf16 [B,H,D,S]
__global__ __launch_bounds__(128) void qkv_rope_kernel(
    const unsigned short* __restrict__ A,     // hidden bf16 [8192,1024]
    const unsigned short* __restrict__ W,     // qkv_w bf16 [3072,1024]
    unsigned short* __restrict__ qbuf,
    unsigned short* __restrict__ kbuf,
    unsigned short* __restrict__ vtbuf) {
    __shared__ __align__(16) unsigned short ldsW[2 * 256 * 32];
    const int lid = threadIdx.x & 31;
    const int wid = threadIdx.x >> 5;
    const int half = lid >> 4, lrow = lid & 15;
    const int n_block = blockIdx.x * 256;     // [0,3072)
    const int n_base = n_block + wid * 64;
    const int m_base = blockIdx.y * 64;       // [0,8192)

    v8f acc[4][4] = {};
    gemm_block_tile(A, W, HID, 3 * HID, m_base, n_block, wid, lid, ldsW, acc);

    const int sel = n_base / HID;             // 0=q 1=k 2=v (tile never straddles)
    const int h   = (n_base & (HID - 1)) >> 6; // head index, tile covers full head
#pragma unroll
    for (int i = 0; i < 4; ++i) {
#pragma unroll
        for (int r = 0; r < 8; ++r) {
            const int row = m_base + 16 * i + half * 8 + r;
            const int b = row >> 11;          // /2048
            const int s = row & (S_LEN - 1);
            if (sel == 2) {                   // V: store transposed [B,H,D,S]
#pragma unroll
                for (int j = 0; j < 4; ++j) {
                    const int d = 16 * j + lrow;
                    vtbuf[((size_t)(b * HEADS + h) * HDIM + d) * S_LEN + s] =
                        f2bf(acc[i][j][r]);
                }
            } else {                          // Q/K: fused RoPE
                unsigned short* dst = (sel == 0) ? qbuf : kbuf;
                const size_t base = ((size_t)(b * HEADS + h) * S_LEN + s) * HDIM;
#pragma unroll
                for (int j = 0; j < 2; ++j) {
                    const int d1 = 16 * j + lrow;           // 0..31
                    // inv_freq = 10000^(-d1/32) ; ln(10000)/32 = 0.28782313662f
                    const float inv = __expf(-(float)d1 * 0.28782313662f);
                    float sn, cs;
                    __sincosf((float)s * inv, &sn, &cs);
                    const float x1 = acc[i][j][r];
                    const float x2 = acc[i][j + 2][r];
                    dst[base + d1]      = f2bf(x1 * cs - x2 * sn);
                    dst[base + d1 + 32] = f2bf(x2 * cs + x1 * sn);
                }
            }
        }
    }
}

// ---------- flash attention: one wave = 16 queries x full S keys ----------
__global__ __launch_bounds__(32) void flash_attn_kernel(
    const unsigned short* __restrict__ qbuf,   // [B,H,S,D] bf16 (RoPE'd)
    const unsigned short* __restrict__ kbuf,   // [B,H,S,D] bf16 (RoPE'd)
    const unsigned short* __restrict__ vtbuf,  // [B,H,D,S] bf16
    unsigned short* __restrict__ attn) {       // [B,S,H*D] bf16
    __shared__ __align__(16) unsigned short pst[16 * 32];  // P tile staging

    const int lid = threadIdx.x;
    const int half = lid >> 4, lrow = lid & 15;
    const int blk = blockIdx.x;
    const int qt = blk & 127;                 // query tile in [0,128)
    const int bh = blk >> 7;                  // b*16+h in [0,64)

    const unsigned short* qp = qbuf + (size_t)bh * S_LEN * HDIM + (size_t)(qt * 16) * HDIM;
    const unsigned short* kp = kbuf + (size_t)bh * S_LEN * HDIM;
    const unsigned short* vp = vtbuf + (size_t)bh * HDIM * S_LEN;

    // Q fragments: two d-slices of 32
    Frag qf[2];
#pragma unroll
    for (int ks = 0; ks < 2; ++ks) {
        const unsigned short* qa = qp + (size_t)lrow * HDIM + ks * 32 + half * 8;
        qf[ks].u[0] = *(const uint4*)(qa);
        qf[ks].u[1] = *(const uint4*)(qa + 16);
    }

    v8f o[4] = {};
    float mi[8], li[8];
#pragma unroll
    for (int r = 0; r < 8; ++r) { mi[r] = -1e30f; li[r] = 0.0f; }

    for (int kb = 0; kb < S_LEN; kb += 32) {
        // ---- scores: Q(16x64) . K^T(64x32) -> two 16x16 f32 tiles ----
        v8f sc[2] = {};
#pragma unroll
        for (int jj = 0; jj < 2; ++jj) {
#pragma unroll
            for (int ks = 0; ks < 2; ++ks) {
                Frag bfr;
                const unsigned short* ka =
                    kp + (size_t)(kb + 16 * jj + lrow) * HDIM + ks * 32 + half * 8;
                bfr.u[0] = *(const uint4*)(ka);
                bfr.u[1] = *(const uint4*)(ka + 16);
                sc[jj] = wmma_bf16(qf[ks].v, bfr.v, sc[jj]);
            }
        }
        // ---- online softmax (rows split across wave halves) ----
        float corr[8];
#pragma unroll
        for (int r = 0; r < 8; ++r) {
            const float a0 = sc[0][r] * 0.125f;    // 1/sqrt(64)
            const float a1 = sc[1][r] * 0.125f;
            float mx = rmax16(fmaxf(a0, a1));
            const float mnew = fmaxf(mi[r], mx);
            const float p0 = __expf(a0 - mnew);
            const float p1 = __expf(a1 - mnew);
            const float c = __expf(mi[r] - mnew);
            const float rs = rsum16(p0 + p1);
            li[r] = li[r] * c + rs;
            mi[r] = mnew;
            corr[r] = c;
            pst[(half * 8 + r) * 32 + lrow]      = f2bf(p0);
            pst[(half * 8 + r) * 32 + 16 + lrow] = f2bf(p1);
        }
        __syncthreads();
        // reload P in A-fragment layout (16x32 bf16)
        Frag pf;
        pf.u[0] = *(const uint4*)&pst[lrow * 32 + half * 8];
        pf.u[1] = *(const uint4*)&pst[lrow * 32 + half * 8 + 16];
        // rescale running output, then accumulate P . V
#pragma unroll
        for (int j = 0; j < 4; ++j)
#pragma unroll
            for (int r = 0; r < 8; ++r) o[j][r] *= corr[r];
#pragma unroll
        for (int j = 0; j < 4; ++j) {
            Frag vfr;
            const unsigned short* va =
                vp + (size_t)(16 * j + lrow) * S_LEN + kb + half * 8;
            vfr.u[0] = *(const uint4*)(va);
            vfr.u[1] = *(const uint4*)(va + 16);
            o[j] = wmma_bf16(pf.v, vfr.v, o[j]);
        }
        __syncthreads();
    }

    // ---- finalize + store to [B,S,H*D] ----
    const int b = bh >> 4, h = bh & 15;
#pragma unroll
    for (int j = 0; j < 4; ++j)
#pragma unroll
        for (int r = 0; r < 8; ++r) {
            const int row = qt * 16 + half * 8 + r;
            const float v = o[j][r] / li[r];
            attn[(size_t)(b * S_LEN + row) * HID + h * HDIM + 16 * j + lrow] = f2bf(v);
        }
}

// ---------- O projection GEMM: out_f32 = attn_bf16 . o_w_bf16^T ----------
__global__ __launch_bounds__(128) void oproj_kernel(
    const unsigned short* __restrict__ A,   // attn bf16 [8192,1024]
    const unsigned short* __restrict__ W,   // o_w  bf16 [1024,1024]
    float* __restrict__ out) {
    __shared__ __align__(16) unsigned short ldsW[2 * 256 * 32];
    const int lid = threadIdx.x & 31;
    const int wid = threadIdx.x >> 5;
    const int half = lid >> 4, lrow = lid & 15;
    const int n_block = blockIdx.x * 256;
    const int n_base = n_block + wid * 64;
    const int m_base = blockIdx.y * 64;

    v8f acc[4][4] = {};
    gemm_block_tile(A, W, HID, HID, m_base, n_block, wid, lid, ldsW, acc);

#pragma unroll
    for (int i = 0; i < 4; ++i)
#pragma unroll
        for (int j = 0; j < 4; ++j)
#pragma unroll
            for (int r = 0; r < 8; ++r) {
                const int row = m_base + 16 * i + half * 8 + r;
                const int col = n_base + 16 * j + lrow;
                out[(size_t)row * HID + col] = acc[i][j][r];
            }
}

// ---------- host launch ----------
extern "C" void kernel_launch(void* const* d_in, const int* in_sizes, int n_in,
                              void* d_out, int out_size, void* d_ws, size_t ws_size,
                              hipStream_t stream) {
    (void)in_sizes; (void)n_in; (void)out_size; (void)ws_size;
    const float* hs   = (const float*)d_in[0];   // [4,2048,1024]
    const float* qkvw = (const float*)d_in[1];   // [3072,1024]
    const float* ow   = (const float*)d_in[2];   // [1024,1024]
    float* out = (float*)d_out;

    char* ws = (char*)d_ws;
    const size_t MB = 1ull << 20;
    unsigned short* hsb   = (unsigned short*)(ws + 0 * MB);   // 16 MB
    unsigned short* qkvwb = (unsigned short*)(ws + 16 * MB);  //  6 MB
    unsigned short* owb   = (unsigned short*)(ws + 22 * MB);  //  2 MB
    unsigned short* qbuf  = (unsigned short*)(ws + 24 * MB);  // 16 MB
    unsigned short* kbuf  = (unsigned short*)(ws + 40 * MB);  // 16 MB
    unsigned short* vtbuf = (unsigned short*)(ws + 56 * MB);  // 16 MB
    unsigned short* attnb = (unsigned short*)(ws + 72 * MB);  // 16 MB

    cvt_bf16_kernel<<<2048, 256, 0, stream>>>(hs, hsb, MROWS * HID);
    cvt_bf16_kernel<<<1024, 256, 0, stream>>>(qkvw, qkvwb, 3 * HID * HID);
    cvt_bf16_kernel<<<512, 256, 0, stream>>>(ow, owb, HID * HID);

    dim3 gq(3 * HID / 256, MROWS / 64);   // (12,128)
    qkv_rope_kernel<<<gq, 128, 0, stream>>>(hsb, qkvwb, qbuf, kbuf, vtbuf);

    flash_attn_kernel<<<BATCH * HEADS * (S_LEN / 16), 32, 0, stream>>>(
        qbuf, kbuf, vtbuf, attnb);

    dim3 go(HID / 256, MROWS / 64);       // (4,128)
    oproj_kernel<<<go, 128, 0, stream>>>(attnb, owb, out);
}
